// DynamicConvolution_45973329937313
// MI455X (gfx1250) — matrix operand order repaired and checked by reading, not compiled, not run on hardware
//
#include <hip/hip_runtime.h>

// Involution-style dynamic depthwise conv, fully fused:
//   H1 = relu(W1 @ X)         (per-pixel 64x64, bf16 WMMA, f32 acc)
//   KERN = W2 @ H1            (per-pixel 576x64, bf16 WMMA, f32 acc, k-major reorder)
//   out[c,p] = sum_k KERN[k,c,p] * x_pad[c, p+shift_k]
// kern is never materialized in HBM. W2 staged in LDS as bf16 (81KB -- uses
// CDNA5's 320KB WGP LDS) to keep VGPRs < 256 (no s_set_vgpr_msb churn).

#define BB 4
#define CC 64
#define HH 224
#define WW 224
#define HWP (HH * WW)
#define TPB 256

typedef __attribute__((ext_vector_type(16))) __bf16 v16bf;
typedef __attribute__((ext_vector_type(2)))  __bf16 v2bf;
typedef __attribute__((ext_vector_type(8)))  float  v8f;

union BfFrag {
  v16bf v;
  unsigned u[8];
};

// one v_cvt_pk_bf16_f32 per pair
__device__ __forceinline__ unsigned pack2(float lo, float hi) {
  v2bf p = {(__bf16)lo, (__bf16)hi};
  return __builtin_bit_cast(unsigned, p);
}

// W2 LDS layout: 576 rows (row = c_out*9 + k), each row = 64 bf16 = 8 uint4,
// padded to 9 uint4 (144B) to spread banks.  Total 576*9 uint4 = 81KB.
#define W2_ROW_Q 9
#define W2_QUADS (576 * W2_ROW_Q)
#define H1_QUADS (2 * 16 * 8)   // two pixel-tile groups x 16 pixels x 128B

__global__ __launch_bounds__(TPB)
void involution_fused_kernel(const float* __restrict__ x,
                             const float* __restrict__ w1,
                             const float* __restrict__ w2,
                             float* __restrict__ out) {
  const int h    = blockIdx.x;        // image row
  const int b    = blockIdx.y;        // batch
  const int tid  = threadIdx.x;
  const int lane = tid & 31;          // wave32 lane
  const int wv2  = tid >> 5;          // wave id 0..7
  const int grp  = wv2 >> 2;          // pixel-tile group 0/1
  const int wv   = wv2 & 3;           // channel group 0..3
  const int g    = lane >> 4;         // lane half (K/M split per ISA layout)
  const int lm   = lane & 15;         // M (A frag) / N (B,C frags) index

  extern __shared__ uint4 smem[];
  uint4* lds_w2 = smem;               // [576][9]
  uint4* lds_h1 = smem + W2_QUADS;    // [2][16][8]  (group, pixel, c/8)

  // ---- stage W2 as bf16 into LDS (cooperative, once per block) ----
  for (int i = tid; i < 576 * 8; i += TPB) {
    const int row = i >> 3, c8 = i & 7;
    const float* src = w2 + row * CC + c8 * 8;
    uint4 q;
    q.x = pack2(src[0], src[1]);
    q.y = pack2(src[2], src[3]);
    q.z = pack2(src[4], src[5]);
    q.w = pack2(src[6], src[7]);
    lds_w2[row * W2_ROW_Q + c8] = q;
  }

  // ---- W1 A-fragments in registers (16 VGPRs) ----
  // 16-bit A 16x32 layout: element e holds K = (e>>3)*16 + g*8 + (e&7), M = lm
  v16bf a1[2];
#pragma unroll
  for (int s = 0; s < 2; ++s) {
#pragma unroll
    for (int e = 0; e < 16; ++e) {
      const int kk = 32 * s + ((e >> 3) * 16) + g * 8 + (e & 7);
      a1[s][e] = (__bf16)w1[(16 * wv + lm) * CC + kk];
    }
  }

  const float* xb = x   + (size_t)b * CC * HWP;
  float*       ob = out + (size_t)b * CC * HWP;

  __syncthreads();  // W2 staged

  for (int wt = 0; wt < WW / 32; ++wt) {
    const int w0 = wt * 32 + grp * 16;   // this group's 16-pixel tile

    // ---- B-fragment of X (K = input channel, N = pixel) ----
    // 16-bit B 32x16 layout: element e holds K = (e>>3)*16 + g*8 + (e&7), N = lm
    v16bf bx[2];
#pragma unroll
    for (int s = 0; s < 2; ++s) {
#pragma unroll
      for (int e = 0; e < 16; ++e) {
        const int c = 32 * s + ((e >> 3) * 16) + g * 8 + (e & 7);
        bx[s][e] = (__bf16)xb[(size_t)c * HWP + h * WW + w0 + lm];
      }
    }
    if (wt + 1 < WW / 32) {
      // pull next tile's x columns toward the caches (global_prefetch_b8)
      __builtin_prefetch(&xb[(size_t)(g * 16) * HWP + h * WW + w0 + 32 + lm], 0, 1);
    }

    // ---- GEMM1: H1 = relu(W1 @ X), wave wv computes rows 16*wv..16*wv+15 ----
    v8f hacc = {};
    hacc = __builtin_amdgcn_wmma_f32_16x16x32_bf16(false, a1[0], false, bx[0],
                                                   (short)0, hacc, false, false);
    hacc = __builtin_amdgcn_wmma_f32_16x16x32_bf16(false, a1[1], false, bx[1],
                                                   (short)0, hacc, false, false);

    __syncthreads();  // previous iteration's lds_h1 readers done
    // C/D f32 layout: VGPR r, lane -> (M = r + 8*g, N = lm)
    // lane owns 8 consecutive channels c = 16*wv + 8*g + r -> one b128 store
    {
      uint4 q;
      q.x = pack2(fmaxf(hacc[0], 0.0f), fmaxf(hacc[1], 0.0f));
      q.y = pack2(fmaxf(hacc[2], 0.0f), fmaxf(hacc[3], 0.0f));
      q.z = pack2(fmaxf(hacc[4], 0.0f), fmaxf(hacc[5], 0.0f));
      q.w = pack2(fmaxf(hacc[6], 0.0f), fmaxf(hacc[7], 0.0f));
      lds_h1[(grp * 16 + lm) * 8 + 2 * wv + g] = q;
    }
    __syncthreads();

    // ---- B-fragment of H1 (K = hidden channel, N = pixel) ----
    // element e -> c = 32*s + (e>>3)*16 + 8*g + (e&7) : two 8-channel runs -> 2x ds_load_b128
    v16bf bh[2];
#pragma unroll
    for (int s = 0; s < 2; ++s) {
      const uint4 q0 = lds_h1[(grp * 16 + lm) * 8 + 4 * s + g];
      const uint4 q1 = lds_h1[(grp * 16 + lm) * 8 + 4 * s + g + 2];
      BfFrag f;
      f.u[0] = q0.x; f.u[1] = q0.y; f.u[2] = q0.z; f.u[3] = q0.w;
      f.u[4] = q1.x; f.u[5] = q1.y; f.u[6] = q1.z; f.u[7] = q1.w;
      bh[s] = f.v;
    }

    // ---- GEMM2 tap-by-tap (A-frags streamed from LDS), fused apply ----
    float oacc[8];
#pragma unroll
    for (int r = 0; r < 8; ++r) oacc[r] = 0.0f;

#pragma unroll
    for (int k = 0; k < 9; ++k) {
      const int dh = k / 3 - 1;
      const int dw = k % 3 - 1;
      // A-fragment for tap k: row = c_out*9 + k, c_out = 16*wv + lm.
      // element e -> kk = 32s + (e>>3)*16 + 8g + (e&7): two 16B runs per kstep.
      const int rowq = ((16 * wv + lm) * 9 + k) * W2_ROW_Q;
      v16bf a2k[2];
#pragma unroll
      for (int s = 0; s < 2; ++s) {
        const uint4 q0 = lds_w2[rowq + 4 * s + g];
        const uint4 q1 = lds_w2[rowq + 4 * s + g + 2];
        BfFrag f;
        f.u[0] = q0.x; f.u[1] = q0.y; f.u[2] = q0.z; f.u[3] = q0.w;
        f.u[4] = q1.x; f.u[5] = q1.y; f.u[6] = q1.z; f.u[7] = q1.w;
        a2k[s] = f.v;
      }
      v8f kacc = {};
      kacc = __builtin_amdgcn_wmma_f32_16x16x32_bf16(false, a2k[0], false, bh[0],
                                                     (short)0, kacc, false, false);
      kacc = __builtin_amdgcn_wmma_f32_16x16x32_bf16(false, a2k[1], false, bh[1],
                                                     (short)0, kacc, false, false);
      const int hh   = h + dh;
      const int wcol = w0 + lm + dw;
      const bool inb = ((unsigned)hh < (unsigned)HH) && ((unsigned)wcol < (unsigned)WW);
#pragma unroll
      for (int r = 0; r < 8; ++r) {
        const int c = 16 * wv + r + 8 * g;
        const float xv = inb ? xb[(size_t)c * HWP + hh * WW + wcol] : 0.0f;
        oacc[r] += kacc[r] * xv;
      }
    }

    // ---- store out[b, c, h, w0+lm] ----
#pragma unroll
    for (int r = 0; r < 8; ++r) {
      const int c = 16 * wv + r + 8 * g;
      ob[(size_t)c * HWP + h * WW + w0 + lm] = oacc[r];
    }
  }
}

extern "C" void kernel_launch(void* const* d_in, const int* in_sizes, int n_in,
                              void* d_out, int out_size, void* d_ws, size_t ws_size,
                              hipStream_t stream) {
  (void)in_sizes; (void)n_in; (void)out_size; (void)d_ws; (void)ws_size;
  const float* x  = (const float*)d_in[0];
  const float* w1 = (const float*)d_in[1];
  const float* w2 = (const float*)d_in[2];
  float* out = (float*)d_out;

  dim3 grid(HH, BB);                  // one block per (row, batch)
  dim3 block(TPB);                    // 8 wave32 waves, two pixel tiles
  const size_t shmem = (size_t)(W2_QUADS + H1_QUADS) * sizeof(uint4);  // 87KB
  involution_fused_kernel<<<grid, block, shmem, stream>>>(x, w1, w2, out);
}
